// MoELayer_27470610825613
// MI455X (gfx1250) — matrix operand order repaired and checked by reading, not compiled, not run on hardware
//
#include <hip/hip_runtime.h>

// ---------------- problem constants ----------------
#define C_DIM   256
#define E_NUM   8
#define H_DIM   682
#define H_PAD   704          // 44 n-tiles of 16 == 22 K-chunks of 32
#define M_TILE  32
#define THREADS 256
#define WAVES   (THREADS / 32)
#define W_ELEMS ((size_t)E_NUM * H_PAD * C_DIM)   // elems per split weight tensor

typedef __attribute__((ext_vector_type(16))) __bf16 v16bf;
typedef __attribute__((ext_vector_type(8)))  __bf16 v8bf;
typedef __attribute__((ext_vector_type(4)))  __bf16 v4bf;
typedef __attribute__((ext_vector_type(8)))  float  v8f;

static __device__ __forceinline__ v8f wmma_bf16(v16bf a, v16bf b, v8f c) {
  // D = A(16x32 bf16) * B(32x16 bf16) + C(16x16 f32)
  return __builtin_amdgcn_wmma_f32_16x16x32_bf16(false, a, false, b, (short)0, c,
                                                 false, false);
}

// A operand (16x32 bf16) for one 16-row tile from row-major LDS (K contiguous).
// Lane layout: lane row M=lane&15 holds K = sel*16..sel*16+15 (matches B)
// where sel = lane>>4.
static __device__ __forceinline__ v16bf lds_loadA(const __bf16* base, int row,
                                                  int stride, int k0, int sel) {
  const v8bf* p0 = (const v8bf*)(base + row * stride + k0 + sel * 16);
  const v8bf* p1 = (const v8bf*)(base + row * stride + k0 + sel * 16 + 8);
  v8bf lo = *p0;
  v8bf hi = *p1;
  return __builtin_shufflevector(lo, hi, 0, 1, 2, 3, 4, 5, 6, 7,
                                 8, 9, 10, 11, 12, 13, 14, 15);
}

// B operand (32x16 bf16): lane n holds 16 consecutive K of its weight row.
// Pre-split bf16 weights are contiguous + 16B aligned -> two global_load_b128.
static __device__ __forceinline__ v16bf g_loadB(const __bf16* p) {
  v8bf a = *(const v8bf*)p;
  v8bf b = *(const v8bf*)(p + 8);
  return __builtin_shufflevector(a, b, 0, 1, 2, 3, 4, 5, 6, 7,
                                 8, 9, 10, 11, 12, 13, 14, 15);
}

static __device__ __forceinline__ void split4(const float f[4], v4bf& ph, v4bf& pl) {
#pragma unroll
  for (int j = 0; j < 4; ++j) {
    __bf16 h = (__bf16)f[j];
    ph[j] = h;
    pl[j] = (__bf16)(f[j] - (float)h);
  }
}

// ---------------- weight pre-split kernels (run once per launch) ----------------
// w1/w3: [E][H_DIM][C] fp32 -> [E][H_PAD][C] bf16 hi/lo (pad rows = 0)
__global__ void __launch_bounds__(THREADS)
split_w_hc(const float* __restrict__ src, __bf16* __restrict__ dh,
           __bf16* __restrict__ dl) {
  const int idx = blockIdx.x * THREADS + threadIdx.x;
  const int total = E_NUM * H_PAD * (C_DIM / 4);
  if (idx >= total) return;
  const int c4 = idx % (C_DIM / 4);
  const int rem = idx / (C_DIM / 4);
  const int hp = rem % H_PAD;
  const int e = rem / H_PAD;
  float4 v = make_float4(0.0f, 0.0f, 0.0f, 0.0f);
  if (hp < H_DIM)
    v = ((const float4*)src)[((size_t)e * H_DIM + hp) * (C_DIM / 4) + c4];
  float f[4] = {v.x, v.y, v.z, v.w};
  v4bf ph, pl;
  split4(f, ph, pl);
  const size_t o = ((size_t)e * H_PAD + hp) * C_DIM + c4 * 4;
  *(v4bf*)(dh + o) = ph;
  *(v4bf*)(dl + o) = pl;
}

// w2: [E][C][H_DIM] fp32 -> [E][C][H_PAD] bf16 hi/lo (pad cols = 0)
__global__ void __launch_bounds__(THREADS)
split_w_ch(const float* __restrict__ src, __bf16* __restrict__ dh,
           __bf16* __restrict__ dl) {
  const int idx = blockIdx.x * THREADS + threadIdx.x;
  const int total = E_NUM * C_DIM * (H_PAD / 4);
  if (idx >= total) return;
  const int h4 = idx % (H_PAD / 4);
  const int rem = idx / (H_PAD / 4);
  const int c = rem % C_DIM;
  const int e = rem / C_DIM;
  const float* sp = src + ((size_t)e * C_DIM + c) * H_DIM;
  float f[4];
#pragma unroll
  for (int j = 0; j < 4; ++j) {
    const int h = h4 * 4 + j;
    f[j] = (h < H_DIM) ? sp[h] : 0.0f;
  }
  v4bf ph, pl;
  split4(f, ph, pl);
  const size_t o = ((size_t)e * C_DIM + c) * H_PAD + h4 * 4;
  *(v4bf*)(dh + o) = ph;
  *(v4bf*)(dl + o) = pl;
}

// ---------------- router: fp32 logits, softmax, top-2, bucket append ----------------
__global__ void __launch_bounds__(THREADS)
router_kernel(const float* __restrict__ x, const float* __restrict__ rw,
              int* __restrict__ cnt, int* __restrict__ tok,
              float* __restrict__ gat, int nTok) {
  __shared__ float srw[E_NUM * C_DIM];
  const int tid = threadIdx.x;
  for (int i = tid; i < E_NUM * C_DIM; i += THREADS) srw[i] = rw[i];
  __syncthreads();

  const int n = blockIdx.x * THREADS + tid;
  if (n >= nTok) return;

  const float4* xr = (const float4*)(x + (size_t)n * C_DIM);
  const float4* w4 = (const float4*)srw;
  float acc[E_NUM];
#pragma unroll
  for (int e = 0; e < E_NUM; ++e) acc[e] = 0.0f;
  for (int c = 0; c < C_DIM / 4; ++c) {
    float4 xv = xr[c];
#pragma unroll
    for (int e = 0; e < E_NUM; ++e) {
      float4 wv = w4[e * (C_DIM / 4) + c];
      acc[e] += xv.x * wv.x + xv.y * wv.y + xv.z * wv.z + xv.w * wv.w;
    }
  }
  float mx = acc[0];
#pragma unroll
  for (int e = 1; e < E_NUM; ++e) mx = fmaxf(mx, acc[e]);
  float p[E_NUM];
  float s = 0.0f;
#pragma unroll
  for (int e = 0; e < E_NUM; ++e) { p[e] = expf(acc[e] - mx); s += p[e]; }

  int e1 = 0; float v1 = p[0];
#pragma unroll
  for (int e = 1; e < E_NUM; ++e)
    if (p[e] > v1) { v1 = p[e]; e1 = e; }        // strict > == lowest-index tiebreak
  int e2 = -1; float v2 = -1.0f;
#pragma unroll
  for (int e = 0; e < E_NUM; ++e)
    if (e != e1 && p[e] > v2) { v2 = p[e]; e2 = e; }

  const float invs = 1.0f / s;
  const float p1 = v1 * invs, p2 = v2 * invs;
  const float rsum = p1 + p2 + 1e-9f;
  const float g1 = p1 / rsum, g2 = p2 / rsum;

  int s1 = atomicAdd(&cnt[e1], 1);
  tok[(size_t)e1 * nTok + s1] = n;
  gat[(size_t)e1 * nTok + s1] = g1;
  int s2 = atomicAdd(&cnt[e2], 1);
  tok[(size_t)e2 * nTok + s2] = n;
  gat[(size_t)e2 * nTok + s2] = g2;
}

// ---------------- expert: SwiGLU FFN over a 32-token tile, bf16x3 WMMA ----------------
__global__ void __launch_bounds__(THREADS)
expert_kernel(const float* __restrict__ x,
              const __bf16* __restrict__ w1h, const __bf16* __restrict__ w1l,
              const __bf16* __restrict__ w3h, const __bf16* __restrict__ w3l,
              const __bf16* __restrict__ w2h, const __bf16* __restrict__ w2l,
              const int* __restrict__ cnt, const int* __restrict__ tok,
              const float* __restrict__ gat, float* __restrict__ out, int nTok) {
  const int e = blockIdx.y;
  const int base = blockIdx.x * M_TILE;
  const int ce = cnt[e];
  if (base >= ce) return;

  extern __shared__ char smem_raw[];
  __bf16* xhi = (__bf16*)smem_raw;               // [32][256]
  __bf16* xlo = xhi + M_TILE * C_DIM;            // [32][256]
  __bf16* hhi = xlo + M_TILE * C_DIM;            // [32][704]
  __bf16* hlo = hhi + M_TILE * H_PAD;            // [32][704]
  int*    stok = (int*)(hlo + M_TILE * H_PAD);   // [32]
  float*  sgat = (float*)(stok + M_TILE);        // [32]

  const int tid = threadIdx.x;
  const int wave = tid >> 5;
  const int lane = tid & 31;
  const int ln = lane & 15;
  const int sel = lane >> 4;

  if (tid < M_TILE) {
    const int slot = base + tid;
    const bool ok = slot < ce;
    stok[tid] = ok ? tok[(size_t)e * nTok + slot] : -1;
    sgat[tid] = ok ? gat[(size_t)e * nTok + slot] : 0.0f;
  }
  __syncthreads();

  // gather X rows, split fp32 -> bf16 hi + residual lo
  for (int i = tid; i < M_TILE * (C_DIM / 4); i += THREADS) {
    const int row = i / (C_DIM / 4);
    const int c4 = i % (C_DIM / 4);
    const int t = stok[row];
    float4 v = make_float4(0.0f, 0.0f, 0.0f, 0.0f);
    if (t >= 0) v = ((const float4*)x)[(size_t)t * (C_DIM / 4) + c4];
    float f[4] = {v.x, v.y, v.z, v.w};
    v4bf ph, pl;
    split4(f, ph, pl);
    *(v4bf*)(xhi + row * C_DIM + c4 * 4) = ph;
    *(v4bf*)(xlo + row * C_DIM + c4 * 4) = pl;
  }
  __syncthreads();

  // ---- stage 1: h = silu(x@w1^T) * (x@w3^T), stored bf16 hi/lo in LDS ----
  for (int ht = wave; ht < H_PAD / 16; ht += WAVES) {
    const int h0 = ht * 16;
    const int hrow = h0 + ln;                    // padded rows are pre-zeroed
    const __bf16* pw1h = w1h + ((size_t)e * H_PAD + hrow) * C_DIM + sel * 16;
    const __bf16* pw1l = w1l + ((size_t)e * H_PAD + hrow) * C_DIM + sel * 16;
    const __bf16* pw3h = w3h + ((size_t)e * H_PAD + hrow) * C_DIM + sel * 16;
    const __bf16* pw3l = w3l + ((size_t)e * H_PAD + hrow) * C_DIM + sel * 16;
    v8f a1m0 = {0, 0, 0, 0, 0, 0, 0, 0}, a1m1 = a1m0, a3m0 = a1m0, a3m1 = a1m0;
#pragma unroll
    for (int kk = 0; kk < C_DIM / 32; ++kk) {
      const v16bf b1h = g_loadB(pw1h + kk * 32);
      const v16bf b1l = g_loadB(pw1l + kk * 32);
      const v16bf b3h = g_loadB(pw3h + kk * 32);
      const v16bf b3l = g_loadB(pw3l + kk * 32);
      const v16bf ah0 = lds_loadA(xhi, ln, C_DIM, kk * 32, sel);
      const v16bf al0 = lds_loadA(xlo, ln, C_DIM, kk * 32, sel);
      const v16bf ah1 = lds_loadA(xhi, ln + 16, C_DIM, kk * 32, sel);
      const v16bf al1 = lds_loadA(xlo, ln + 16, C_DIM, kk * 32, sel);
      a1m0 = wmma_bf16(ah0, b1h, a1m0);
      a1m0 = wmma_bf16(ah0, b1l, a1m0);
      a1m0 = wmma_bf16(al0, b1h, a1m0);
      a1m1 = wmma_bf16(ah1, b1h, a1m1);
      a1m1 = wmma_bf16(ah1, b1l, a1m1);
      a1m1 = wmma_bf16(al1, b1h, a1m1);
      a3m0 = wmma_bf16(ah0, b3h, a3m0);
      a3m0 = wmma_bf16(ah0, b3l, a3m0);
      a3m0 = wmma_bf16(al0, b3h, a3m0);
      a3m1 = wmma_bf16(ah1, b3h, a3m1);
      a3m1 = wmma_bf16(ah1, b3l, a3m1);
      a3m1 = wmma_bf16(al1, b3h, a3m1);
    }
    // C/D layout: vgpr r, lane l -> M = r + 8*(l>>4), N = l&15
#pragma unroll
    for (int r = 0; r < 8; ++r) {
      const int m = r + 8 * sel;
      {
        const float g = a1m0[r];
        const float hv = (g / (1.0f + __expf(-g))) * a3m0[r];
        const __bf16 hh = (__bf16)hv;
        hhi[m * H_PAD + h0 + ln] = hh;
        hlo[m * H_PAD + h0 + ln] = (__bf16)(hv - (float)hh);
      }
      {
        const float g = a1m1[r];
        const float hv = (g / (1.0f + __expf(-g))) * a3m1[r];
        const __bf16 hh = (__bf16)hv;
        hhi[(m + 16) * H_PAD + h0 + ln] = hh;
        hlo[(m + 16) * H_PAD + h0 + ln] = (__bf16)(hv - (float)hh);
      }
    }
  }
  __syncthreads();

  // ---- stage 2: y = h @ w2^T ; out[token] += gate * y ----
  for (int ct = wave; ct < C_DIM / 16; ct += WAVES) {
    const int ccol = ct * 16 + ln;
    const __bf16* pw2h = w2h + ((size_t)e * C_DIM + ccol) * H_PAD + sel * 16;
    const __bf16* pw2l = w2l + ((size_t)e * C_DIM + ccol) * H_PAD + sel * 16;
    v8f ym0 = {0, 0, 0, 0, 0, 0, 0, 0}, ym1 = ym0;
    for (int kk = 0; kk < H_PAD / 32; ++kk) {
      const v16bf bh = g_loadB(pw2h + kk * 32);
      const v16bf bl = g_loadB(pw2l + kk * 32);
      const v16bf ah0 = lds_loadA(hhi, ln, H_PAD, kk * 32, sel);
      const v16bf al0 = lds_loadA(hlo, ln, H_PAD, kk * 32, sel);
      const v16bf ah1 = lds_loadA(hhi, ln + 16, H_PAD, kk * 32, sel);
      const v16bf al1 = lds_loadA(hlo, ln + 16, H_PAD, kk * 32, sel);
      ym0 = wmma_bf16(ah0, bh, ym0);
      ym0 = wmma_bf16(ah0, bl, ym0);
      ym0 = wmma_bf16(al0, bh, ym0);
      ym1 = wmma_bf16(ah1, bh, ym1);
      ym1 = wmma_bf16(ah1, bl, ym1);
      ym1 = wmma_bf16(al1, bh, ym1);
    }
#pragma unroll
    for (int r = 0; r < 8; ++r) {
      const int m = r + 8 * sel;
      const int t0 = stok[m];
      if (t0 >= 0)
        atomicAdd(out + (size_t)t0 * C_DIM + ct * 16 + ln, sgat[m] * ym0[r]);
      const int t1 = stok[m + 16];
      if (t1 >= 0)
        atomicAdd(out + (size_t)t1 * C_DIM + ct * 16 + ln, sgat[m + 16] * ym1[r]);
    }
  }
}

extern "C" void kernel_launch(void* const* d_in, const int* in_sizes, int n_in,
                              void* d_out, int out_size, void* d_ws, size_t ws_size,
                              hipStream_t stream) {
  const float* x = (const float*)d_in[0];
  const float* rw = (const float*)d_in[1];
  const float* w1 = (const float*)d_in[2];
  const float* w2 = (const float*)d_in[3];
  const float* w3 = (const float*)d_in[4];
  float* out = (float*)d_out;

  const int nTok = in_sizes[0] / C_DIM;   // B*T = 16384

  // workspace layout:
  //   [cnt: 8 int, padded to 256B] [tok: E*N int] [gat: E*N f32]
  //   [w1h|w1l|w3h|w3l: E*H_PAD*C bf16 each] [w2h|w2l: E*C*H_PAD bf16 each]
  char* ws = (char*)d_ws;
  int* cnt = (int*)ws;
  int* tok = (int*)(ws + 256);
  float* gat = (float*)(ws + 256 + (size_t)E_NUM * nTok * sizeof(int));
  size_t woff = 256 + (size_t)E_NUM * nTok * (sizeof(int) + sizeof(float));
  woff = (woff + 255) & ~(size_t)255;
  __bf16* w1h = (__bf16*)(ws + woff);
  __bf16* w1l = w1h + W_ELEMS;
  __bf16* w3h = w1l + W_ELEMS;
  __bf16* w3l = w3h + W_ELEMS;
  __bf16* w2h = w3l + W_ELEMS;
  __bf16* w2l = w2h + W_ELEMS;

  hipMemsetAsync(d_ws, 0, 256, stream);
  hipMemsetAsync(d_out, 0, (size_t)out_size * sizeof(float), stream);

  router_kernel<<<(nTok + THREADS - 1) / THREADS, THREADS, 0, stream>>>(
      x, rw, cnt, tok, gat, nTok);

  const int nsplit = E_NUM * H_PAD * (C_DIM / 4);          // == E*C*(H_PAD/4)
  const int sgrid = (nsplit + THREADS - 1) / THREADS;
  split_w_hc<<<sgrid, THREADS, 0, stream>>>(w1, w1h, w1l);
  split_w_hc<<<sgrid, THREADS, 0, stream>>>(w3, w3h, w3l);
  split_w_ch<<<sgrid, THREADS, 0, stream>>>(w2, w2h, w2l);

  dim3 grid((nTok + M_TILE - 1) / M_TILE, E_NUM);
  const size_t smem =
      (size_t)(2 * M_TILE * C_DIM + 2 * M_TILE * H_PAD) * sizeof(__bf16) +
      M_TILE * sizeof(int) + M_TILE * sizeof(float);   // 123,136 B (<320KB WGP LDS)
  expert_kernel<<<grid, THREADS, smem, stream>>>(x, w1h, w1l, w3h, w3l, w2h, w2l,
                                                 cnt, tok, gat, out, nTok);
}
